// GC_LSTM_36601711297144
// MI455X (gfx1250) — matrix operand order repaired and checked by reading, not compiled, not run on hardware
//
#include <hip/hip_runtime.h>
#include <hip/hip_bf16.h>
#include <math.h>

// ---------------------------------------------------------------------------
// Problem constants
// ---------------------------------------------------------------------------
#define BB   8
#define TT   32
#define CC   4880
#define KC_E 153           // ceil(4880/32) K-chunks for embedding (zero-padded)
#define DD   64
#define HH   64
#define AA   256
#define DIN  128           // 2*D
#define MM   256           // B*T rows

typedef _Float16 v16h __attribute__((ext_vector_type(16)));
typedef _Float16 h8   __attribute__((ext_vector_type(8)));
typedef _Float16 h2   __attribute__((ext_vector_type(2)));
typedef float    v8f  __attribute__((ext_vector_type(8)));
typedef float    f4   __attribute__((ext_vector_type(4)));

// ---------------------------------------------------------------------------
// WMMA helpers. Fragments are stored PRE-PERMUTED: lane L's 16 halfs live at
// [chunkBase + 16*L .. +15] -> a fragment load is two b128 loads.
// Permutation (inverse of CDNA5 ISA 7.12.2 wave32 layouts), chunk-local k<32:
//   A (16x32): lane = m + 16*k[3],  j = k[2:0] + 8*k[4]
//   B (32x16): lane = n + 16*k[3],  j = k[2:0] + 8*k[4]
// C/D: element r <-> row (r + 8*(lane>=16)), col (lane&15)
// ---------------------------------------------------------------------------
__device__ __forceinline__ v8f wmma_f16(v16h a, v16h b, v8f c) {
  return __builtin_amdgcn_wmma_f32_16x16x32_f16(
      false, a, false, b, (short)0, c, false, false);
}
__device__ __forceinline__ int a_slot(int m, int k) {
  return ((m + (((k >> 3) & 1) << 4)) << 4) + (k & 7) + (((k >> 4) & 1) << 3);
}
__device__ __forceinline__ v16h frag_ld(const _Float16* p, int lane) {
  return *(const v16h*)(p + lane * 16);
}
__device__ __forceinline__ float sigf(float x) { return 1.f / (1.f + __expf(-x)); }

// ---------------------------------------------------------------------------
// K0: repack f32 matrix W[K][N] into f16 fragment-ordered chunks with zero
// padding up to KC*32 rows:  Wp[(nt*KC + kt)*512 + lane*16 + j]
// ---------------------------------------------------------------------------
__global__ void k_pack_b(const float* __restrict__ W, _Float16* __restrict__ Wp,
                         int K, int KC, int N) {
  int o = blockIdx.x * blockDim.x + threadIdx.x;
  if (o >= (N >> 4) * KC * 512) return;
  int j = o & 15, lane = (o >> 4) & 31, chunk = o >> 9;
  int kt = chunk % KC, nt = chunk / KC;
  int k = kt * 32 + ((lane >> 4) << 3) + (j & 7) + ((j >> 3) << 4);
  int n = nt * 16 + (lane & 15);
  Wp[o] = (_Float16)((k < K) ? W[k * N + n] : 0.f);
}

// ---------------------------------------------------------------------------
// K1: S[256,64] = X[256,4880] @ E[4880,64] + per-row cnt.
// One wave per 16-row M-tile, 4 accumulators (all 64 N cols). Zero LDS, zero
// barriers, branch-free K-loop and tail. A fragments straight from global x
// (4x b128 + cvt_pk), B fragments from packed emb (2x b128 each).
// ---------------------------------------------------------------------------
__global__ __launch_bounds__(32) void k_embed_gemm(
    const float* __restrict__ x, const _Float16* __restrict__ embp,
    float* __restrict__ S, float* __restrict__ cnt) {
  const int lane = threadIdx.x & 31;
  const int m0 = blockIdx.x * 16;
  const int m = lane & 15, hi = (lane >> 4) << 3;
  const float* xr = x + (m0 + m) * CC + hi;     // lane's k window base
  const _Float16* eb = embp + lane * 16;
  float psum = 0.f;
  v8f acc0 = {}, acc1 = {}, acc2 = {}, acc3 = {};
  for (int k0 = 0; k0 < CC - 16; k0 += 32) {    // 152 branch-free iters
    if (k0 < 4800) __builtin_prefetch(xr + k0 + 64, 0, 3);
    f4 x0 = *(const f4*)(xr + k0);
    f4 x1 = *(const f4*)(xr + k0 + 4);
    f4 x2 = *(const f4*)(xr + k0 + 16);
    f4 x3 = *(const f4*)(xr + k0 + 20);
    psum += (x0.x + x0.y + x0.z + x0.w) + (x1.x + x1.y + x1.z + x1.w)
          + (x2.x + x2.y + x2.z + x2.w) + (x3.x + x3.y + x3.z + x3.w);
    v16h a;
    a[0]  = (_Float16)x0.x; a[1]  = (_Float16)x0.y; a[2]  = (_Float16)x0.z; a[3]  = (_Float16)x0.w;
    a[4]  = (_Float16)x1.x; a[5]  = (_Float16)x1.y; a[6]  = (_Float16)x1.z; a[7]  = (_Float16)x1.w;
    a[8]  = (_Float16)x2.x; a[9]  = (_Float16)x2.y; a[10] = (_Float16)x2.z; a[11] = (_Float16)x2.w;
    a[12] = (_Float16)x3.x; a[13] = (_Float16)x3.y; a[14] = (_Float16)x3.z; a[15] = (_Float16)x3.w;
    const int kt = k0 >> 5;
    acc0 = wmma_f16(a, *(const v16h*)(eb + (0 * KC_E + kt) * 512), acc0);
    acc1 = wmma_f16(a, *(const v16h*)(eb + (1 * KC_E + kt) * 512), acc1);
    acc2 = wmma_f16(a, *(const v16h*)(eb + (2 * KC_E + kt) * 512), acc2);
    acc3 = wmma_f16(a, *(const v16h*)(eb + (3 * KC_E + kt) * 512), acc3);
  }
  {                                             // tail: 16 valid cols = elems 0..7
    f4 x0 = *(const f4*)(xr + (CC - 16));
    f4 x1 = *(const f4*)(xr + (CC - 16) + 4);
    psum += (x0.x + x0.y + x0.z + x0.w) + (x1.x + x1.y + x1.z + x1.w);
    v16h a = {};
    a[0] = (_Float16)x0.x; a[1] = (_Float16)x0.y; a[2] = (_Float16)x0.z; a[3] = (_Float16)x0.w;
    a[4] = (_Float16)x1.x; a[5] = (_Float16)x1.y; a[6] = (_Float16)x1.z; a[7] = (_Float16)x1.w;
    const int kt = KC_E - 1;                    // packed emb zero-filled there
    acc0 = wmma_f16(a, *(const v16h*)(eb + (0 * KC_E + kt) * 512), acc0);
    acc1 = wmma_f16(a, *(const v16h*)(eb + (1 * KC_E + kt) * 512), acc1);
    acc2 = wmma_f16(a, *(const v16h*)(eb + (2 * KC_E + kt) * 512), acc2);
    acc3 = wmma_f16(a, *(const v16h*)(eb + (3 * KC_E + kt) * 512), acc3);
  }
  psum += __shfl_xor(psum, 16);                 // lanes m & m+16 cover full row
  if (lane < 16) cnt[m0 + lane] = fmaxf(psum, 1.f);
  const int mh = (lane >> 4) << 3, nl = lane & 15;
#pragma unroll
  for (int r = 0; r < 8; ++r) {
    float* sr = S + (m0 + r + mh) * DD + nl;
    sr[0] = acc0[r]; sr[16] = acc1[r]; sr[32] = acc2[r]; sr[48] = acc3[r];
  }
}

// ---------------------------------------------------------------------------
// K2: avg_xt = S/cnt ; avg_ht = avg_xt @ Wg + bg ; lstm_in = [avg_ht|avg_xt]
// One wave per M-tile, LDS-free: A from S (b128 + scale + cvt), B from packed
// Wg. avg_xt written straight from fragment registers (16B stores).
// ---------------------------------------------------------------------------
__global__ __launch_bounds__(32) void k_cheb(
    const float* __restrict__ S, const float* __restrict__ cnt,
    const _Float16* __restrict__ Wgp, const float* __restrict__ bg,
    _Float16* __restrict__ lstm_in) {
  const int lane = threadIdx.x & 31;
  const int m0 = blockIdx.x * 16;
  const int m = lane & 15, hi = (lane >> 4) << 3;
  const float ic = 1.f / cnt[m0 + m];
  const float* sr = S + (m0 + m) * DD + hi;
  v8f acc0 = {}, acc1 = {}, acc2 = {}, acc3 = {};
#pragma unroll
  for (int kc = 0; kc < 2; ++kc) {
    f4 s0 = *(const f4*)(sr + kc * 32);
    f4 s1 = *(const f4*)(sr + kc * 32 + 4);
    f4 s2 = *(const f4*)(sr + kc * 32 + 16);
    f4 s3 = *(const f4*)(sr + kc * 32 + 20);
    v16h a;
    a[0]  = (_Float16)(s0.x * ic); a[1]  = (_Float16)(s0.y * ic);
    a[2]  = (_Float16)(s0.z * ic); a[3]  = (_Float16)(s0.w * ic);
    a[4]  = (_Float16)(s1.x * ic); a[5]  = (_Float16)(s1.y * ic);
    a[6]  = (_Float16)(s1.z * ic); a[7]  = (_Float16)(s1.w * ic);
    a[8]  = (_Float16)(s2.x * ic); a[9]  = (_Float16)(s2.y * ic);
    a[10] = (_Float16)(s2.z * ic); a[11] = (_Float16)(s2.w * ic);
    a[12] = (_Float16)(s3.x * ic); a[13] = (_Float16)(s3.y * ic);
    a[14] = (_Float16)(s3.z * ic); a[15] = (_Float16)(s3.w * ic);
    // avg_xt half: lane m covers cols kc*32+hi+0..7 and kc*32+16+hi+0..7
    h8 lo = {a[0], a[1], a[2],  a[3],  a[4],  a[5],  a[6],  a[7]};
    h8 hb = {a[8], a[9], a[10], a[11], a[12], a[13], a[14], a[15]};
    *(h8*)(lstm_in + (m0 + m) * DIN + 64 + kc * 32 + hi)      = lo;
    *(h8*)(lstm_in + (m0 + m) * DIN + 64 + kc * 32 + 16 + hi) = hb;
    const _Float16* wb = Wgp + lane * 16 + kc * 512;  // chunks: (nt*2 + kc)*512
    acc0 = wmma_f16(a, *(const v16h*)(wb + 0 * 1024), acc0);
    acc1 = wmma_f16(a, *(const v16h*)(wb + 1 * 1024), acc1);
    acc2 = wmma_f16(a, *(const v16h*)(wb + 2 * 1024), acc2);
    acc3 = wmma_f16(a, *(const v16h*)(wb + 3 * 1024), acc3);
  }
  const int mh = (lane >> 4) << 3, nl = lane & 15;
#pragma unroll
  for (int r = 0; r < 8; ++r) {
    _Float16* lr = lstm_in + (m0 + r + mh) * DIN;
    lr[0  + nl] = (_Float16)(acc0[r] + bg[0  + nl]);
    lr[16 + nl] = (_Float16)(acc1[r] + bg[16 + nl]);
    lr[32 + nl] = (_Float16)(acc2[r] + bg[32 + nl]);
    lr[48 + nl] = (_Float16)(acc3[r] + bg[48 + nl]);
  }
}

// ---------------------------------------------------------------------------
// K3: bidirectional LSTM; one persistent block per direction, 16 waves.
// Packed global weights -> b128 fragment loads; 6 WMMAs/step.
// Hidden states also emitted directly in packed-A layout for k_attn.
// ---------------------------------------------------------------------------
__global__ __launch_bounds__(512) void k_lstm(
    const _Float16* __restrict__ lstm_in,
    const _Float16* __restrict__ Wihp_f, const _Float16* __restrict__ Whhp_f,
    const float* __restrict__ b_f,
    const _Float16* __restrict__ Wihp_b, const _Float16* __restrict__ Whhp_b,
    const float* __restrict__ b_b,
    float* __restrict__ lstm_out, _Float16* __restrict__ att_a) {
  __shared__ __align__(32) _Float16 xt_p[4 * 512];
  __shared__ __align__(32) _Float16 h_p[2 * 512];
  __shared__ float c_s[8 * 64];
  __shared__ float z_s[16 * 256];
  const int dir = blockIdx.x;
  const _Float16* Wih = dir ? Wihp_b : Wihp_f;
  const _Float16* Whh = dir ? Whhp_b : Whhp_f;
  const float*    bia = dir ? b_b : b_f;
  const int tid = threadIdx.x, lane = tid & 31, wid = tid >> 5;
  for (int i = tid; i < 4 * 512; i += 512) xt_p[i] = (_Float16)0.f;  // pads rows 8..15
  for (int i = tid; i < 2 * 512; i += 512) h_p[i]  = (_Float16)0.f;
  for (int i = tid; i < 8 * 64;  i += 512) c_s[i]  = 0.f;
  __syncthreads();
  const int n0 = wid * 16;
  const float bn = bia[n0 + (lane & 15)];
  const int xb = tid >> 6, xk = (tid & 63) * 2;         // xt staging: 2 halfs/thread
  const int xslot = (xk >> 5) * 512 + a_slot(xb, xk & 31);
  const int eb = tid >> 6, ehc = tid & 63;              // elementwise mapping
  const int hslot = (ehc >> 5) * 512 + a_slot(eb, ehc & 31);
  for (int s = 0; s < TT; ++s) {
    const int t = dir ? (TT - 1 - s) : s;
    *(h2*)(xt_p + xslot) = *(const h2*)(lstm_in + (xb * TT + t) * DIN + xk);
    __syncthreads();
    v8f acc;
#pragma unroll
    for (int r = 0; r < 8; ++r) acc[r] = bn;            // bias splat (col-only)
#pragma unroll
    for (int kt = 0; kt < 4; ++kt)                      // xt @ Wih
      acc = wmma_f16(frag_ld(xt_p + kt * 512, lane),
                     frag_ld(Wih + (wid * 4 + kt) * 512, lane), acc);
#pragma unroll
    for (int kt = 0; kt < 2; ++kt)                      // h @ Whh
      acc = wmma_f16(frag_ld(h_p + kt * 512, lane),
                     frag_ld(Whh + (wid * 2 + kt) * 512, lane), acc);
    {
      const int n = n0 + (lane & 15), mh = (lane >> 4) << 3;
#pragma unroll
      for (int r = 0; r < 8; ++r) z_s[(r + mh) * 256 + n] = acc[r];
    }
    __syncthreads();
    {                                                   // gates: 8x64 entries
      float zi = z_s[eb * 256 + ehc];
      float zf = z_s[eb * 256 + 64 + ehc];
      float zg = z_s[eb * 256 + 128 + ehc];
      float zo = z_s[eb * 256 + 192 + ehc];
      float c  = sigf(zf) * c_s[eb * 64 + ehc] + sigf(zi) * tanhf(zg);
      float h  = sigf(zo) * tanhf(c);
      c_s[eb * 64 + ehc] = c;
      h_p[hslot] = (_Float16)h;
      const int m = eb * TT + t, kk = dir * HH + ehc;
      lstm_out[m * (2 * HH) + kk] = h;
      att_a[((m >> 4) * 4 + (kk >> 5)) * 512 + a_slot(m & 15, kk & 31)] = (_Float16)h;
    }
    __syncthreads();
  }
}

// ---------------------------------------------------------------------------
// K4: e[m] = tanh(lstm_out[m,:] @ Wa + ba) . ua — A and B both packed global.
// ---------------------------------------------------------------------------
__global__ __launch_bounds__(512) void k_attn(
    const _Float16* __restrict__ att_a, const _Float16* __restrict__ Wap,
    const float* __restrict__ ba, const float* __restrict__ ua,
    float* __restrict__ e_out) {
  __shared__ float e_s[16];
  const int tid = threadIdx.x, lane = tid & 31, wid = tid >> 5;
  const int mt = blockIdx.x;
  if (tid < 16) e_s[tid] = 0.f;
  __syncthreads();
  const int n = wid * 16 + (lane & 15);
  const float ban = ba[n], uan = ua[n];
  v8f acc;
#pragma unroll
  for (int r = 0; r < 8; ++r) acc[r] = ban;
#pragma unroll
  for (int kt = 0; kt < 4; ++kt)
    acc = wmma_f16(frag_ld(att_a + (mt * 4 + kt) * 512, lane),
                   frag_ld(Wap + (wid * 4 + kt) * 512, lane), acc);
  const int mh = (lane >> 4) << 3;
#pragma unroll
  for (int r = 0; r < 8; ++r)
    atomicAdd(&e_s[r + mh], tanhf(acc[r]) * uan);
  __syncthreads();
  if (tid < 16) e_out[mt * 16 + tid] = e_s[tid];
}

// ---------------------------------------------------------------------------
// K5: per-batch softmax over T (wave32 shuffles), ctx, leaky_relu, head, sigmoid
// ---------------------------------------------------------------------------
__global__ __launch_bounds__(256) void k_final(
    const float* __restrict__ e_in, const float* __restrict__ lstm_out,
    const float* __restrict__ Wo, const float* __restrict__ bo,
    float* __restrict__ out) {
  __shared__ float alpha_s[8 * 32];
  __shared__ float out_s[8];
  const int tid = threadIdx.x, lane = tid & 31, bw = tid >> 5;
  if (tid < 8) out_s[tid] = 0.f;
  float ev = e_in[bw * TT + lane];
  float mx = ev;
#pragma unroll
  for (int off = 16; off > 0; off >>= 1) mx = fmaxf(mx, __shfl_xor(mx, off));
  float ex = __expf(ev - mx);
  float sm = ex;
#pragma unroll
  for (int off = 16; off > 0; off >>= 1) sm += __shfl_xor(sm, off);
  alpha_s[bw * TT + lane] = ex / sm;
  __syncthreads();
#pragma unroll
  for (int i = 0; i < 4; ++i) {                         // 8*128 ctx entries
    int e = i * 256 + tid;
    int b = e >> 7, h = e & 127;
    float s = 0.f;
    for (int t = 0; t < TT; ++t)
      s += alpha_s[b * TT + t] * lstm_out[(b * TT + t) * (2 * HH) + h];
    float v = s > 0.f ? s : 0.5f * s;                   // leaky_relu(0.5)
    atomicAdd(&out_s[b], v * Wo[h]);
  }
  __syncthreads();
  if (tid < 8) out[tid] = 1.f / (1.f + __expf(-(out_s[tid] + bo[0])));
}

// ---------------------------------------------------------------------------
// Host launcher
// ---------------------------------------------------------------------------
extern "C" void kernel_launch(void* const* d_in, const int* in_sizes, int n_in,
                              void* d_out, int out_size, void* d_ws, size_t ws_size,
                              hipStream_t stream) {
  (void)in_sizes; (void)n_in; (void)out_size; (void)ws_size;
  const float* x     = (const float*)d_in[0];
  // d_in[1] = visit_lens (int64) — unused by the reference computation
  const float* emb   = (const float*)d_in[2];
  const float* Wg    = (const float*)d_in[3];
  const float* bg    = (const float*)d_in[4];
  const float* Wih_f = (const float*)d_in[5];
  const float* Whh_f = (const float*)d_in[6];
  const float* b_f   = (const float*)d_in[7];
  const float* Wih_b = (const float*)d_in[8];
  const float* Whh_b = (const float*)d_in[9];
  const float* b_b   = (const float*)d_in[10];
  const float* Wa    = (const float*)d_in[11];
  const float* ba    = (const float*)d_in[12];
  const float* ua    = (const float*)d_in[13];
  const float* Wo    = (const float*)d_in[14];
  const float* bo    = (const float*)d_in[15];
  float* out = (float*)d_out;

  // workspace layout (f32 region then 32B-aligned f16 region)
  float* ws_f        = (float*)d_ws;
  float* S           = ws_f;                 // 16384
  float* cnt         = ws_f + 16384;         // 256
  float* lstm_out    = ws_f + 16640;         // 32768
  float* e_arr       = ws_f + 49408;         // 256
  _Float16* h16      = (_Float16*)(ws_f + 49664);
  _Float16* lstm_in_h = h16;                 // 32768 (row-major [256][128])
  _Float16* att_a     = h16 + 32768;         // 32768 (packed A fragments)
  _Float16* Wihp_f    = h16 + 65536;         // 32768
  _Float16* Whhp_f    = h16 + 98304;         // 16384
  _Float16* Wihp_b    = h16 + 114688;        // 32768
  _Float16* Whhp_b    = h16 + 147456;        // 16384
  _Float16* Wap       = h16 + 163840;        // 32768
  _Float16* Wgp       = h16 + 196608;        // 8192  (Wg packed, KC=2, N=64)
  _Float16* embp      = h16 + 204800;        // 313344 (emb packed, KC=153, N=64)

  // K0: repack weights/embedding into f16 fragment-ordered layout
  k_pack_b<<<(4 * KC_E * 512 + 255) / 256, 256, 0, stream>>>(emb, embp, CC, KC_E, DD);
  k_pack_b<<<(2 * 4  * 512 + 255) / 256, 256, 0, stream>>>(Wg, Wgp, DD, 2, DD);
  k_pack_b<<<(4 * 16 * 512 + 255) / 256, 256, 0, stream>>>(Wih_f, Wihp_f, DIN, 4, 4 * HH);
  k_pack_b<<<(2 * 16 * 512 + 255) / 256, 256, 0, stream>>>(Whh_f, Whhp_f, HH, 2, 4 * HH);
  k_pack_b<<<(4 * 16 * 512 + 255) / 256, 256, 0, stream>>>(Wih_b, Wihp_b, DIN, 4, 4 * HH);
  k_pack_b<<<(2 * 16 * 512 + 255) / 256, 256, 0, stream>>>(Whh_b, Whhp_b, HH, 2, 4 * HH);
  k_pack_b<<<(4 * 16 * 512 + 255) / 256, 256, 0, stream>>>(Wa, Wap, DIN, 4, AA);

  // K1: S = x @ embedding, cnt = row sums (1 wave per M-tile, no LDS/barriers)
  k_embed_gemm<<<MM / 16, 32, 0, stream>>>(x, embp, S, cnt);
  // K2: avg + ChebConv -> lstm_in (f16), LDS-free
  k_cheb<<<MM / 16, 32, 0, stream>>>(S, cnt, Wgp, bg, lstm_in_h);
  // K3: bidirectional LSTM
  k_lstm<<<2, 512, 0, stream>>>(lstm_in_h, Wihp_f, Whhp_f, b_f,
                                Wihp_b, Whhp_b, b_b, lstm_out, att_a);
  // K4: attention scores
  k_attn<<<MM / 16, 512, 0, stream>>>(att_a, Wap, ba, ua, e_arr);
  // K5: softmax + context + head
  k_final<<<1, 256, 0, stream>>>(e_arr, lstm_out, Wo, bo, out);
}
